// SpeechActLSTM_34557306864034
// MI455X (gfx1250) — matrix operand order repaired and checked by reading, not compiled, not run on hardware
//
#include <hip/hip_runtime.h>
#include <hip/hip_bf16.h>

typedef __attribute__((ext_vector_type(16))) __bf16 v16bf;
typedef __attribute__((ext_vector_type(8)))  float  v8f;

#define TT      8192      // total word timesteps (512 utt * 16 words)
#define NU      512       // utterances
#define HH      512       // hidden size
#define GG      2048      // 4*H gate rows
#define KK      512       // inner dim of all GEMMs
#define LBL     32

// ---------------- elementwise / gather kernels ----------------

__global__ void k_embed_bf16(const int* __restrict__ tok,
                             const float* __restrict__ emb,
                             __bf16* __restrict__ out, int n) {
  int i = blockIdx.x * blockDim.x + threadIdx.x;
  if (i < n) {
    int t = i >> 9, k = i & 511;
    out[i] = (__bf16)emb[(size_t)tok[t] * HH + k];
  }
}

__global__ void k_f32_to_bf16(const float* __restrict__ src,
                              __bf16* __restrict__ dst, int n) {
  int i = blockIdx.x * blockDim.x + threadIdx.x;
  if (i < n) dst[i] = (__bf16)src[i];
}

__global__ void k_vadd(const float* __restrict__ a, const float* __restrict__ b,
                       float* __restrict__ o, int n) {
  int i = blockIdx.x * blockDim.x + threadIdx.x;
  if (i < n) o[i] = a[i] + b[i];
}

__global__ void k_uttrep_bf16(const float* __restrict__ h1,
                              __bf16* __restrict__ out, int n) {
  int i = blockIdx.x * blockDim.x + threadIdx.x;
  if (i < n) {
    int u = i >> 9, k = i & 511;
    out[i] = (__bf16)h1[(size_t)((u + 1) * 16 - 1) * HH + k];
  }
}

// ---------------- WMMA GEMM: P[M,N] = A[M,K] @ W[N,K]^T (bf16 in, f32 out) --
// 2x2 register blocking: each wave owns a 32x32 output patch (4 accumulators).
// Per 32-wide K chunk: 2 A-fragment loads + 2 B-fragment loads -> 4 WMMAs,
// i.e. one 32B fragment load per wmma (2x the arithmetic intensity of 1-tile).

__device__ __forceinline__ v16bf load_a_frag(const __bf16* arow, int k0, int aoff) {
  v16bf av;
  uint4* ap = reinterpret_cast<uint4*>(&av);
  ap[0] = *reinterpret_cast<const uint4*>(arow + k0 + aoff);       // k = j + aoff     (j=0..7)
  ap[1] = *reinterpret_cast<const uint4*>(arow + k0 + 16 + aoff);  // k = j + 8 + aoff (j=8..15)
  return av;
}

__device__ __forceinline__ v16bf load_b_frag(const __bf16* brow, int k0, int boff) {
  v16bf bv;
  uint4* bp = reinterpret_cast<uint4*>(&bv);
  bp[0] = *reinterpret_cast<const uint4*>(brow + k0 + boff);       // k = j + boff
  bp[1] = *reinterpret_cast<const uint4*>(brow + k0 + 16 + boff);
  return bv;
}

__global__ void k_wmma_gemm(const __bf16* __restrict__ A,
                            const __bf16* __restrict__ W,
                            float* __restrict__ P,
                            int M, int N, int K) {
  const int lane   = threadIdx.x & 31;
  const int wave   = threadIdx.x >> 5;
  const int tilesN = N >> 5;                       // 32-wide patches in N
  const int tile   = blockIdx.x * (blockDim.x >> 5) + wave;
  const int numTiles = (M >> 5) * tilesN;
  if (tile >= numTiles) return;                    // wave-uniform: EXEC all-ones

  const int tm   = (tile / tilesN) << 5;
  const int tn   = (tile % tilesN) << 5;
  const int half = lane >> 4;                      // 0: lanes 0-15, 1: lanes 16-31
  const int l15  = lane & 15;
  const int aoff = half * 8;                       // A k-offset (ISA 16-bit A layout)
  const int boff = half * 8;                       // B k-offset (same k order as A)

  const __bf16* arow0 = A + (size_t)(tm + l15) * K;
  const __bf16* arow1 = arow0 + (size_t)16 * K;
  const __bf16* brow0 = W + (size_t)(tn + l15) * K;
  const __bf16* brow1 = brow0 + (size_t)16 * K;

  v8f acc00 = {}, acc01 = {}, acc10 = {}, acc11 = {};
  for (int k0 = 0; k0 < K; k0 += 32) {
    v16bf a0 = load_a_frag(arow0, k0, aoff);
    v16bf a1 = load_a_frag(arow1, k0, aoff);
    v16bf b0 = load_b_frag(brow0, k0, boff);
    v16bf b1 = load_b_frag(brow1, k0, boff);
    if (k0 + 32 < K) {
      __builtin_prefetch(arow0 + k0 + 32, 0, 1);
      __builtin_prefetch(brow0 + k0 + 32, 0, 1);
    }
    acc00 = __builtin_amdgcn_wmma_f32_16x16x32_bf16(false, a0, false, b0, (short)0, acc00, false, false);
    acc10 = __builtin_amdgcn_wmma_f32_16x16x32_bf16(false, a1, false, b0, (short)0, acc10, false, false);
    acc01 = __builtin_amdgcn_wmma_f32_16x16x32_bf16(false, a0, false, b1, (short)0, acc01, false, false);
    acc11 = __builtin_amdgcn_wmma_f32_16x16x32_bf16(false, a1, false, b1, (short)0, acc11, false, false);
  }
  // C/D layout: element j -> row (sub_tm + j + 8*half), col (sub_tn + (lane&15))
#pragma unroll
  for (int j = 0; j < 8; ++j) {
    const int m0 = tm + j + half * 8;
    P[(size_t)m0 * N + tn + l15]             = acc00[j];
    P[(size_t)m0 * N + tn + 16 + l15]        = acc01[j];
    P[(size_t)(m0 + 16) * N + tn + l15]      = acc10[j];
    P[(size_t)(m0 + 16) * N + tn + 16 + l15] = acc11[j];
  }
}

// ---------------- persistent grid-synced LSTM recurrence -------------------
// 32 WGs x 256 threads. WG wg owns hidden units [wg*16, wg*16+16) => 64 gate
// rows resident in LDS (stride 513: bank-conflict-free since 513 % 64 == 1).
// Per step: all WGs read full h_{t-1} from global, matvec their LDS slice,
// apply gate nonlinearities for their 16 units, write h_t, grid-barrier.

#define RG 32
#define RT 256
#define WSTRIDE 513

__global__ void k_lstm_scan(const float* __restrict__ pre,   // [T, 2048] x@Wih^T
                            const float* __restrict__ whh,   // [2048, 512]
                            const float* __restrict__ bias,  // [2048] b_ih+b_hh
                            float* __restrict__ hout,        // [T, 512]
                            int T, unsigned int* counter) {
  extern __shared__ float sm[];
  float* wsl  = sm;                      // 64 * 513
  float* hsh  = wsl + 64 * WSTRIDE;      // 512
  float* part = hsh + HH;                // 256
  float* gval = part + RT;               // 64
  float* cst  = gval + 64;               // 16

  const int tid = threadIdx.x;
  const int wg  = blockIdx.x;            // 0..31
  const int j0  = wg * 16;

  // Load this WG's W_hh slice into LDS once (row r = g*16+u, g in {i,f,g,o}).
  for (int idx = tid; idx < 64 * HH; idx += RT) {
    int r = idx / HH, c = idx - r * HH;
    int g = r >> 4, u = r & 15;
    wsl[r * WSTRIDE + c] = whh[(size_t)(g * HH + j0 + u) * HH + c];
  }
  if (tid < 16) cst[tid] = 0.0f;

  const int r  = tid & 63;               // local gate row for reduce stage
  const int p  = tid >> 6;               // quarter of the dot product
  const int gw = (r >> 4) * HH + j0 + (r & 15);   // global gate row
  const float bb = bias[gw];
  __syncthreads();

  for (int t = 0; t < T; ++t) {
    if (t > 0) {
      if (tid == 0) {
        const unsigned tgt = (unsigned)(RG * t);
        while (__hip_atomic_load(counter, __ATOMIC_ACQUIRE,
                                 __HIP_MEMORY_SCOPE_AGENT) < tgt)
          __builtin_amdgcn_s_sleep(1);
      }
      __syncthreads();
      __threadfence();                   // acquire: make other WGs' h_t-1 visible
      for (int i = tid; i < HH; i += RT) hsh[i] = hout[(size_t)(t - 1) * HH + i];
    } else {
      for (int i = tid; i < HH; i += RT) hsh[i] = 0.0f;
    }
    __syncthreads();

    // 4 threads per gate row, 128 MACs each (weights LDS, h broadcast-read)
    float s = 0.0f;
    const float* wr = &wsl[r * WSTRIDE + p * 128];
    const float* hr = &hsh[p * 128];
#pragma unroll 8
    for (int i = 0; i < 128; ++i) s += wr[i] * hr[i];
    part[tid] = s;
    __syncthreads();

    if (tid < 64) {
      float d = part[tid] + part[tid + 64] + part[tid + 128] + part[tid + 192];
      gval[tid] = d + pre[(size_t)t * GG + gw] + bb;
    }
    __syncthreads();

    if (tid < 16) {
      float iv = gval[tid],      fv = gval[16 + tid];
      float gv = gval[32 + tid], ov = gval[48 + tid];
      iv = 1.0f / (1.0f + expf(-iv));
      fv = 1.0f / (1.0f + expf(-fv));
      ov = 1.0f / (1.0f + expf(-ov));
      float c_ = fv * cst[tid] + iv * tanhf(gv);
      cst[tid] = c_;
      hout[(size_t)t * HH + j0 + tid] = ov * tanhf(c_);
    }
    __threadfence();                     // release h_t before arriving
    __syncthreads();
    if (tid == 0)
      __hip_atomic_fetch_add(counter, 1u, __ATOMIC_RELEASE,
                             __HIP_MEMORY_SCOPE_AGENT);
  }
}

// ---------------- decoder (exact fp32, tiny) -------------------------------

__global__ void k_decoder(const float* __restrict__ hu,
                          const float* __restrict__ dw,
                          const float* __restrict__ db,
                          float* __restrict__ em) {
  int i = blockIdx.x * blockDim.x + threadIdx.x;   // NU*LBL = 16384
  if (i < NU * LBL) {
    int t = i >> 5, l = i & 31;
    const float* h = hu + (size_t)t * HH;
    const float* w = dw + (size_t)l * HH;
    float s = db[l];
    for (int k = 0; k < HH; ++k) s += h[k] * w[k];
    em[i] = s;
  }
}

// ---------------- CRF log-likelihood (one wave32) --------------------------

__global__ void k_crf(const float* __restrict__ em, const int* __restrict__ tgt,
                      const float* __restrict__ cs, const float* __restrict__ ce,
                      const float* __restrict__ ct, float* __restrict__ out) {
  __shared__ float trans[LBL * LBL];
  __shared__ float alpha[LBL];
  __shared__ float red[LBL];
  const int j = threadIdx.x;
  for (int i = j; i < LBL * LBL; i += LBL) trans[i] = ct[i];
  __syncthreads();

  float a = cs[j] + em[j];
  for (int t = 1; t < NU; ++t) {
    alpha[j] = a;
    __syncthreads();
    float m = -1e30f;
    for (int i = 0; i < LBL; ++i) m = fmaxf(m, alpha[i] + trans[i * LBL + j]);
    float s = 0.0f;
    for (int i = 0; i < LBL; ++i) s += expf(alpha[i] + trans[i * LBL + j] - m);
    a = m + logf(s) + em[t * LBL + j];
    __syncthreads();
  }
  alpha[j] = a + ce[j];
  __syncthreads();
  float m = -1e30f;
  for (int i = 0; i < LBL; ++i) m = fmaxf(m, alpha[i]);
  float s = 0.0f;
  for (int i = 0; i < LBL; ++i) s += expf(alpha[i] - m);
  const float den = m + logf(s);

  float num = 0.0f;
  for (int t = j; t < NU; t += LBL) num += em[t * LBL + tgt[t]];
  for (int t = j; t < NU - 1; t += LBL) num += trans[tgt[t] * LBL + tgt[t + 1]];
  red[j] = num;
  __syncthreads();
  if (j == 0) {
    float ns = cs[tgt[0]] + ce[tgt[NU - 1]];
    for (int i = 0; i < LBL; ++i) ns += red[i];
    out[0] = -((ns - den) / (float)NU);
  }
}

// ---------------- host-side orchestration ----------------------------------

extern "C" void kernel_launch(void* const* d_in, const int* in_sizes, int n_in,
                              void* d_out, int out_size, void* d_ws, size_t ws_size,
                              hipStream_t stream) {
  const int*   tokens  = (const int*)d_in[0];
  const int*   targets = (const int*)d_in[1];
  const float* emb     = (const float*)d_in[2];
  const float* w_ih    = (const float*)d_in[3];   // [2,2048,512]
  const float* w_hh    = (const float*)d_in[4];   // [2,2048,512]
  const float* b_ih    = (const float*)d_in[5];   // [2,2048]
  const float* b_hh    = (const float*)d_in[6];
  const float* wu_ih   = (const float*)d_in[7];   // [2048,512]
  const float* wu_hh   = (const float*)d_in[8];
  const float* bu_ih   = (const float*)d_in[9];
  const float* bu_hh   = (const float*)d_in[10];
  const float* dec_w   = (const float*)d_in[11];
  const float* dec_b   = (const float*)d_in[12];
  const float* crf_s   = (const float*)d_in[13];
  const float* crf_e   = (const float*)d_in[14];
  const float* crf_t   = (const float*)d_in[15];

  char* W = (char*)d_ws;
  size_t off = 0;
  float*  pre   = (float*)(W + off);  off += (size_t)TT * GG * 4;      // 64 MB (reused for all 3 pre-gate GEMMs)
  __bf16* seqb  = (__bf16*)(W + off); off += (size_t)TT * HH * 2;      // 8 MB
  __bf16* wih0b = (__bf16*)(W + off); off += (size_t)GG * KK * 2;      // 2 MB
  __bf16* wih1b = (__bf16*)(W + off); off += (size_t)GG * KK * 2;
  __bf16* wuihb = (__bf16*)(W + off); off += (size_t)GG * KK * 2;
  float*  hA    = (float*)(W + off);  off += (size_t)TT * HH * 4;      // 16 MB
  float*  hB    = (float*)(W + off);  off += (size_t)TT * HH * 4;      // 16 MB
  __bf16* hbf   = (__bf16*)(W + off); off += (size_t)TT * HH * 2;      // 8 MB (reused for utt_rep)
  float*  outu  = (float*)(W + off);  off += (size_t)NU * HH * 4;      // 1 MB
  float*  emis  = (float*)(W + off);  off += (size_t)NU * LBL * 4;
  float*  bias0 = (float*)(W + off);  off += GG * 4;
  float*  bias1 = (float*)(W + off);  off += GG * 4;
  float*  biasU = (float*)(W + off);  off += GG * 4;
  unsigned* cnt = (unsigned*)(W + off); off += 128;

  const size_t scanLds = (size_t)(64 * WSTRIDE + HH + RT + 64 + 16) * sizeof(float);

  // Prep: gather+convert activations, convert input-projection weights, fuse biases
  k_embed_bf16 <<<TT * HH / 256, 256, 0, stream>>>(tokens, emb, seqb, TT * HH);
  k_f32_to_bf16<<<GG * KK / 256, 256, 0, stream>>>(w_ih,            wih0b, GG * KK);
  k_f32_to_bf16<<<GG * KK / 256, 256, 0, stream>>>(w_ih + (size_t)GG * KK, wih1b, GG * KK);
  k_f32_to_bf16<<<GG * KK / 256, 256, 0, stream>>>(wu_ih,           wuihb, GG * KK);
  k_vadd<<<GG / 256, 256, 0, stream>>>(b_ih,      b_hh,      bias0, GG);
  k_vadd<<<GG / 256, 256, 0, stream>>>(b_ih + GG, b_hh + GG, bias1, GG);
  k_vadd<<<GG / 256, 256, 0, stream>>>(bu_ih,     bu_hh,     biasU, GG);

  // Layer 0: hoisted input GEMM (WMMA, 32x32 patches) + grid-synced recurrence
  k_wmma_gemm<<<(TT / 32) * (GG / 32) / 8, 256, 0, stream>>>(seqb, wih0b, pre, TT, GG, KK);
  hipMemsetAsync(cnt, 0, sizeof(unsigned), stream);
  k_lstm_scan<<<RG, RT, scanLds, stream>>>(pre, w_hh, bias0, hA, TT, cnt);

  // Layer 1
  k_f32_to_bf16<<<TT * HH / 256, 256, 0, stream>>>(hA, hbf, TT * HH);
  k_wmma_gemm<<<(TT / 32) * (GG / 32) / 8, 256, 0, stream>>>(hbf, wih1b, pre, TT, GG, KK);
  hipMemsetAsync(cnt, 0, sizeof(unsigned), stream);
  k_lstm_scan<<<RG, RT, scanLds, stream>>>(pre, w_hh + (size_t)GG * KK, bias1, hB, TT, cnt);

  // Utterance LSTM over last-token states
  k_uttrep_bf16<<<NU * HH / 256, 256, 0, stream>>>(hB, hbf, NU * HH);
  k_wmma_gemm<<<(NU / 32) * (GG / 32) / 8, 256, 0, stream>>>(hbf, wuihb, pre, NU, GG, KK);
  hipMemsetAsync(cnt, 0, sizeof(unsigned), stream);
  k_lstm_scan<<<RG, RT, scanLds, stream>>>(pre, wu_hh, biasU, outu, NU, cnt);

  // Decoder + CRF
  k_decoder<<<NU * LBL / 256, 256, 0, stream>>>(outu, dec_w, dec_b, emis);
  k_crf<<<1, LBL, 0, stream>>>(emis, targets, crf_s, crf_e, crf_t, (float*)d_out);
  (void)in_sizes; (void)n_in; (void)out_size; (void)ws_size;
}
